// Type_aware_GAT_28724741276290
// MI455X (gfx1250) — matrix operand order repaired and checked by reading, not compiled, not run on hardware
//
#include <hip/hip_runtime.h>

// Type-aware GAT (Trans_Gat_Layer x2 heads + Muti_GAT fusion) for gfx1250.
// Strategy: algebraic collapse of the repeat_interleave attention-score tensor
// and of the edge-MLP (V*H lookup table), then bf16 WMMA (16x16x32, f32 acc)
// for the four remaining GEMM groups.

#define ALPHA 0.2f
#define NEGV  -9e15f

constexpr int Bb = 8, Nn = 256, Dd = 256, Hh = 2, Vv = 50, Ee = 100;

typedef float  v8f   __attribute__((ext_vector_type(8)));
typedef __bf16 v8bf  __attribute__((ext_vector_type(8)));
typedef __bf16 v16bf __attribute__((ext_vector_type(16)));

__device__ __forceinline__ float lrelu(float v) { return v > 0.f ? v : ALPHA * v; }

// A fragment (16x32 bf16, row-major A, lda in elements).
// Lane m = lane&15; koff = (lane>>4)*16. Elements e = A[m][kb+koff+e]
// (contiguous k order, matching load_bt_frag).
__device__ __forceinline__ v16bf load_a_frag(const __bf16* A, int lda, int m0, int kb, int lane) {
  int m = m0 + (lane & 15);
  int koff = (lane >> 4) * 16;
  const __bf16* p = A + (size_t)m * lda + kb + koff;
  v8bf lo = *(const v8bf*)p;
  v8bf hi = *(const v8bf*)(p + 8);
  return __builtin_shufflevector(lo, hi, 0,1,2,3,4,5,6,7,8,9,10,11,12,13,14,15);
}

// B fragment (32x16 bf16) from a pre-transposed Bt[n][k] (ldb in elements).
// Lane n = lane&15; koff2 = (lane>>4)*16. Elements e = B[kb+koff2+e][n],
// contiguous 32B in Bt.
__device__ __forceinline__ v16bf load_bt_frag(const __bf16* Bt, int ldb, int n0, int kb, int lane) {
  int n = n0 + (lane & 15);
  int koff2 = (lane >> 4) * 16;
  return *(const v16bf*)(Bt + (size_t)n * ldb + kb + koff2);
}

__device__ __forceinline__ v8f wmma_bf16(v16bf a, v16bf b, v8f c) {
  return __builtin_amdgcn_wmma_f32_16x16x32_bf16(false, a, false, b, (short)0, c, false, false);
}

// C/D layout: VGPR v: lanes 0-15 -> M=v, lanes 16-31 -> M=v+8; N = lane&15.
__device__ __forceinline__ void store_tile(float* C, int ldc, int m0, int n0, int lane, v8f acc) {
  int col = lane & 15;
  int mb  = (lane >> 4) * 8;
#pragma unroll
  for (int v = 0; v < 8; ++v)
    C[(size_t)(m0 + mb + v) * ldc + n0 + col] = acc[v];
}

// ---------------- staging kernels ----------------

__global__ void cvt_lin_kernel(const float* __restrict__ src, __bf16* __restrict__ dst, int n) {
  int i = blockIdx.x * blockDim.x + threadIdx.x;
  if (i < n) dst[i] = (__bf16)src[i];
}

// dst[batch][c][r] = (bf16) src[batch][r][c]; R,C multiples of 16.
__global__ void transpose_cvt_kernel(const float* __restrict__ src, __bf16* __restrict__ dst,
                                     int R, int C) {
  __shared__ float t[16][17];
  size_t mat = (size_t)blockIdx.z * R * C;
  int c0 = blockIdx.x * 16, r0 = blockIdx.y * 16;
  t[threadIdx.y][threadIdx.x] = src[mat + (size_t)(r0 + threadIdx.y) * C + c0 + threadIdx.x];
  __syncthreads();
  dst[mat + (size_t)(c0 + threadIdx.y) * R + r0 + threadIdx.x] = (__bf16)t[threadIdx.x][threadIdx.y];
}

// ---------------- GEMM 1: h = x@W, xet = x@et^T ----------------

__global__ void gemm_xw_kernel(const __bf16* __restrict__ x_bf,
                               const __bf16* __restrict__ Wt,    // [H][d][k] = W[h][k][d]
                               const __bf16* __restrict__ etT,   // [H][d][k] = et_w[h][d][k]
                               float* __restrict__ h_out,
                               float* __restrict__ xet_out) {
  int h = blockIdx.z, b = blockIdx.y;
  int wave = threadIdx.x >> 5, lane = threadIdx.x & 31;
  int tile = blockIdx.x * 8 + wave;          // 0..255
  int m0 = (tile >> 4) * 16, n0 = (tile & 15) * 16;
  const __bf16* A  = x_bf + (size_t)b * Nn * Dd;
  const __bf16* B1 = Wt  + (size_t)h * Dd * Dd;
  const __bf16* B2 = etT + (size_t)h * Dd * Dd;
  v8f acc1 = {}; v8f acc2 = {};
  for (int kb = 0; kb < Dd; kb += 32) {
    v16bf a  = load_a_frag(A, Dd, m0, kb, lane);
    v16bf b1 = load_bt_frag(B1, Dd, n0, kb, lane);
    v16bf b2 = load_bt_frag(B2, Dd, n0, kb, lane);
    acc1 = wmma_bf16(a, b1, acc1);
    acc2 = wmma_bf16(a, b2, acc2);
  }
  store_tile(h_out   + (size_t)(h * Bb + b) * Nn * Dd, Dd, m0, n0, lane, acc1);
  store_tile(xet_out + (size_t)(h * Bb + b) * Nn * Dd, Dd, m0, n0, lane, acc2);
}

// ---------------- small precompute kernels ----------------

__global__ void s_left_kernel(const float* __restrict__ a, float* __restrict__ s_left) {
  __shared__ float sm[256];
  int h = blockIdx.x, t = threadIdx.x;
  sm[t] = a[(size_t)h * 2 * Dd + t];         // t < D
  __syncthreads();
  for (int s = 128; s > 0; s >>= 1) { if (t < s) sm[t] += sm[t + s]; __syncthreads(); }
  if (t == 0) s_left[h] = sm[0];
}

__global__ void score_right_kernel(const float* __restrict__ hmat, const float* __restrict__ a,
                                   float* __restrict__ score_r) {
  int hb = blockIdx.x;                        // h*B + b
  int h = hb / Bb;
  int n = threadIdx.x;
  const float* ar = a + (size_t)h * 2 * Dd + Dd;
  const float* hr = hmat + ((size_t)hb * Nn + n) * Dd;
  float s = 0.f;
  for (int d = 0; d < Dd; ++d) s += hr[d] * ar[d];
  score_r[(size_t)hb * Nn + n] = s;
}

// ew_val[h][v] = leaky( relu(edge_table[v]@em_w[h]+em_b[h]) @ ew_w[h] + ew_b[h] )
__global__ void ew_val_kernel(const float* __restrict__ edge_table,
                              const float* __restrict__ em_w, const float* __restrict__ em_b,
                              const float* __restrict__ ew_w, const float* __restrict__ ew_b,
                              float* __restrict__ ew_val) {
  __shared__ float sm[256];
  int v = blockIdx.x, h = blockIdx.y, d = threadIdx.x;
  float s = em_b[(size_t)h * Dd + d];
  for (int e = 0; e < Ee; ++e)
    s += edge_table[(size_t)v * Ee + e] * em_w[((size_t)h * Ee + e) * Dd + d];
  s = fmaxf(s, 0.f);
  sm[d] = s * ew_w[(size_t)h * Dd + d];
  __syncthreads();
  for (int st = 128; st > 0; st >>= 1) { if (d < st) sm[d] += sm[d + st]; __syncthreads(); }
  if (d == 0) ew_val[h * Vv + v] = lrelu(sm[0] + ew_b[h]);
}

// ---------------- masked softmax (both heads, both branches) ----------------

__global__ void attn_softmax_kernel(const float* __restrict__ hmat,
                                    const float* __restrict__ score_r,
                                    const float* __restrict__ s_left,
                                    const float* __restrict__ ew_val,
                                    const int* __restrict__ adj,
                                    const int* __restrict__ dep,
                                    float* __restrict__ attT_f, float* __restrict__ attE_f,
                                    __bf16* __restrict__ attT_b, __bf16* __restrict__ attE_b) {
  __shared__ float sm[256];
  int row = blockIdx.x;                       // b*N + i
  int b = row >> 8, i = row & 255;
  int j = threadIdx.x;
  int aj = adj[(size_t)row * Nn + j];
  int dv = dep[(size_t)row * Nn + j];
  for (int h = 0; h < Hh; ++h) {
    size_t hb = (size_t)h * Bb + b;
    size_t base = (hb * Nn + i) * Nn;
    // ---- text branch: e = leaky(h[b,i,j]*S_left + h[b,j,:].a_right) ----
    float hv = hmat[(hb * Nn + i) * Dd + j];  // D==N: j indexes the feature dim
    float e = lrelu(hv * s_left[h] + score_r[hb * Nn + j]);
    float et = (aj > 0) ? e : NEGV;
    sm[j] = et; __syncthreads();
    for (int s = 128; s > 0; s >>= 1) { if (j < s) sm[j] = fmaxf(sm[j], sm[j + s]); __syncthreads(); }
    float mx = sm[0]; __syncthreads();
    float ex = expf(et - mx);
    sm[j] = ex; __syncthreads();
    for (int s = 128; s > 0; s >>= 1) { if (j < s) sm[j] += sm[j + s]; __syncthreads(); }
    float av = ex / sm[0]; __syncthreads();
    attT_f[base + j] = av;
    attT_b[base + j] = (__bf16)av;
    // ---- edge branch: table lookup ----
    float ee = (aj > 0) ? ew_val[h * Vv + dv] : NEGV;
    sm[j] = ee; __syncthreads();
    for (int s = 128; s > 0; s >>= 1) { if (j < s) sm[j] = fmaxf(sm[j], sm[j + s]); __syncthreads(); }
    float mx2 = sm[0]; __syncthreads();
    float ex2 = expf(ee - mx2);
    sm[j] = ex2; __syncthreads();
    for (int s = 128; s > 0; s >>= 1) { if (j < s) sm[j] += sm[j + s]; __syncthreads(); }
    float av2 = ex2 / sm[0]; __syncthreads();
    attE_f[base + j] = av2;
    attE_b[base + j] = (__bf16)av2;
  }
}

// ---------------- GEMM 2: head-mean of att_t@h and att_e@xet ----------------

__global__ void gemm_attn_kernel(const __bf16* __restrict__ attT,
                                 const __bf16* __restrict__ attE,
                                 const __bf16* __restrict__ hT,    // [h*B+b][d][j]
                                 const __bf16* __restrict__ xetT,  // [h*B+b][d][j]
                                 float* __restrict__ htm, float* __restrict__ hem) {
  int b = blockIdx.y;
  int wave = threadIdx.x >> 5, lane = threadIdx.x & 31;
  int tile = blockIdx.x * 8 + wave;
  int m0 = (tile >> 4) * 16, n0 = (tile & 15) * 16;
  v8f at = {}; v8f ae = {};
  for (int h = 0; h < Hh; ++h) {
    const __bf16* At = attT + ((size_t)h * Bb + b) * Nn * Nn;
    const __bf16* Ae = attE + ((size_t)h * Bb + b) * Nn * Nn;
    const __bf16* Bt = hT   + ((size_t)h * Bb + b) * Dd * Nn;
    const __bf16* Be = xetT + ((size_t)h * Bb + b) * Dd * Nn;
    for (int kb = 0; kb < Nn; kb += 32) {
      v16bf a1 = load_a_frag(At, Nn, m0, kb, lane);
      v16bf a2 = load_a_frag(Ae, Nn, m0, kb, lane);
      v16bf b1 = load_bt_frag(Bt, Nn, n0, kb, lane);
      v16bf b2 = load_bt_frag(Be, Nn, n0, kb, lane);
      at = wmma_bf16(a1, b1, at);
      ae = wmma_bf16(a2, b2, ae);
    }
  }
  const float s = 1.0f / (float)Hh;
  at = at * s; ae = ae * s;
  store_tile(htm + (size_t)b * Nn * Dd, Dd, m0, n0, lane, at);
  store_tile(hem + (size_t)b * Nn * Dd, Dd, m0, n0, lane, ae);
}

// ---------------- gated fusion + outputs ----------------

__global__ void fuse_kernel(const float* __restrict__ htm, const float* __restrict__ hem,
                            const float* __restrict__ attT_f, const float* __restrict__ attE_f,
                            const float* __restrict__ ts_w, const float* __restrict__ ts_b,
                            const float* __restrict__ es_w, const float* __restrict__ es_b,
                            float* __restrict__ out) {
  __shared__ float sm[256];
  int row = blockIdx.x;                       // b*N + n
  int d = threadIdx.x;
  float ht = htm[(size_t)row * Dd + d];
  float he = hem[(size_t)row * Dd + d];
  // ch = [h_edge, h_text]
  sm[d] = he * ts_w[d] + ht * ts_w[Dd + d]; __syncthreads();
  for (int s = 128; s > 0; s >>= 1) { if (d < s) sm[d] += sm[d + s]; __syncthreads(); }
  float st_ = sm[0]; __syncthreads();
  sm[d] = he * es_w[d] + ht * es_w[Dd + d]; __syncthreads();
  for (int s = 128; s > 0; s >>= 1) { if (d < s) sm[d] += sm[d + s]; __syncthreads(); }
  float se_ = sm[0]; __syncthreads();
  float sig_t = 1.f / (1.f + expf(-(st_ + ts_b[0])));
  float sig_e = 1.f / (1.f + expf(-(se_ + es_b[0])));
  out[(size_t)row * Dd + d] = fmaxf(sig_t * ht + (1.f - sig_t) * he, 0.f);
  // attention output (head-means), j == d
  float atm = 0.f, aem = 0.f;
  for (int h = 0; h < Hh; ++h) {
    size_t base = ((size_t)h * Bb * Nn + row) * Nn;
    atm += attT_f[base + d];
    aem += attE_f[base + d];
  }
  atm *= (1.f / (float)Hh); aem *= (1.f / (float)Hh);
  out[(size_t)Bb * Nn * Dd + (size_t)row * Nn + d] = sig_t * atm + sig_e * aem;
}

extern "C" void kernel_launch(void* const* d_in, const int* in_sizes, int n_in,
                              void* d_out, int out_size, void* d_ws, size_t ws_size,
                              hipStream_t stream) {
  (void)in_sizes; (void)n_in; (void)out_size; (void)ws_size;
  const float* x          = (const float*)d_in[0];
  const int*   adj        = (const int*)d_in[1];
  const int*   dep        = (const int*)d_in[2];
  const float* edge_table = (const float*)d_in[3];
  const float* W          = (const float*)d_in[4];
  const float* a          = (const float*)d_in[5];
  const float* em_w       = (const float*)d_in[6];
  const float* em_b       = (const float*)d_in[7];
  const float* ew_w       = (const float*)d_in[8];
  const float* ew_b       = (const float*)d_in[9];
  const float* et_w       = (const float*)d_in[10];
  const float* ts_w       = (const float*)d_in[11];
  const float* ts_b       = (const float*)d_in[12];
  const float* es_w       = (const float*)d_in[13];
  const float* es_b       = (const float*)d_in[14];
  float* out = (float*)d_out;

  char* p = (char*)d_ws;
  auto alloc = [&](size_t bytes) -> void* {
    void* r = (void*)p;
    p += (bytes + 255) & ~(size_t)255;
    return r;
  };
  __bf16* x_bf   = (__bf16*)alloc((size_t)Bb * Nn * Dd * 2);
  __bf16* Wt_bf  = (__bf16*)alloc((size_t)Hh * Dd * Dd * 2);
  __bf16* etT_bf = (__bf16*)alloc((size_t)Hh * Dd * Dd * 2);
  float*  h_f    = (float*)alloc((size_t)Hh * Bb * Nn * Dd * 4);
  float*  xet_f  = (float*)alloc((size_t)Hh * Bb * Nn * Dd * 4);
  __bf16* hT     = (__bf16*)alloc((size_t)Hh * Bb * Dd * Nn * 2);
  __bf16* xetT   = (__bf16*)alloc((size_t)Hh * Bb * Dd * Nn * 2);
  float*  attT_f = (float*)alloc((size_t)Hh * Bb * Nn * Nn * 4);
  float*  attE_f = (float*)alloc((size_t)Hh * Bb * Nn * Nn * 4);
  __bf16* attT_b = (__bf16*)alloc((size_t)Hh * Bb * Nn * Nn * 2);
  __bf16* attE_b = (__bf16*)alloc((size_t)Hh * Bb * Nn * Nn * 2);
  float*  htm    = (float*)alloc((size_t)Bb * Nn * Dd * 4);
  float*  hem    = (float*)alloc((size_t)Bb * Nn * Dd * 4);
  float*  score_r= (float*)alloc((size_t)Hh * Bb * Nn * 4);
  float*  s_left = (float*)alloc((size_t)Hh * 4);
  float*  ew_val = (float*)alloc((size_t)Hh * Vv * 4);

  // Stage bf16 operands.
  cvt_lin_kernel<<<(Bb * Nn * Dd + 255) / 256, 256, 0, stream>>>(x, x_bf, Bb * Nn * Dd);
  transpose_cvt_kernel<<<dim3(Dd / 16, Dd / 16, Hh), dim3(16, 16), 0, stream>>>(W, Wt_bf, Dd, Dd);
  cvt_lin_kernel<<<(Hh * Dd * Dd + 255) / 256, 256, 0, stream>>>(et_w, etT_bf, Hh * Dd * Dd);

  // h = x@W, xet = x@et^T (WMMA).
  gemm_xw_kernel<<<dim3(32, Bb, Hh), 256, 0, stream>>>(x_bf, Wt_bf, etT_bf, h_f, xet_f);

  // Transposed bf16 copies for the attention GEMM B-operands.
  transpose_cvt_kernel<<<dim3(Dd / 16, Nn / 16, Hh * Bb), dim3(16, 16), 0, stream>>>(h_f, hT, Nn, Dd);
  transpose_cvt_kernel<<<dim3(Dd / 16, Nn / 16, Hh * Bb), dim3(16, 16), 0, stream>>>(xet_f, xetT, Nn, Dd);

  // Scalar precomputes for the collapsed attention score + edge table.
  s_left_kernel<<<Hh, 256, 0, stream>>>(a, s_left);
  score_right_kernel<<<Hh * Bb, 256, 0, stream>>>(h_f, a, score_r);
  ew_val_kernel<<<dim3(Vv, Hh), 256, 0, stream>>>(edge_table, em_w, em_b, ew_w, ew_b, ew_val);

  // Masked softmaxes for both heads/branches.
  attn_softmax_kernel<<<Bb * Nn, 256, 0, stream>>>(h_f, score_r, s_left, ew_val, adj, dep,
                                                   attT_f, attE_f, attT_b, attE_b);

  // Head-mean att@h and att@xet (WMMA).
  gemm_attn_kernel<<<dim3(32, Bb), 256, 0, stream>>>(attT_b, attE_b, hT, xetT, htm, hem);

  // Gated fusion + both outputs.
  fuse_kernel<<<Bb * Nn, 256, 0, stream>>>(htm, hem, attT_f, attE_f, ts_w, ts_b, es_w, es_b, out);
}
